// DyRABlock_10780367912997
// MI455X (gfx1250) — matrix operand order repaired and checked by reading, not compiled, not run on hardware
//
#include <hip/hip_runtime.h>

// ---------------------------------------------------------------------------
// DyRABlock for MI455X (gfx1250): bf16 WMMA GEMM convs + bandwidth kernels.
// Templated GEMM: compile-time shapes -> no runtime int division, unrolled
// multi-tile WMMA per barrier, prefetch of packed A stream.
// ---------------------------------------------------------------------------

typedef __bf16 bf16_t;
typedef __attribute__((ext_vector_type(16))) __bf16 v16bf;
typedef __attribute__((ext_vector_type(8)))  float  v8f;

// Problem constants (match setup_inputs()).
constexpr int B_  = 8;
constexpr int C_  = 256;
constexpr int H_  = 56;
constexpr int W_  = 56;
constexpr int HW_ = H_ * W_;
constexpr int NPIX = B_ * HW_;          // 25088
constexpr int HID  = 64;                // max(32, C/4)
constexpr int R_   = 2;
constexpr int OFFC = 18;                // 2*3*3 offset channels
constexpr int KD   = 7;                 // pooled size
constexpr float EPS_BN = 1e-5f;

__device__ __forceinline__ bf16_t f2bf(float f) {
  union { float f; unsigned u; } a; a.f = f;
  unsigned u = a.u;
  unsigned r = u + 0x7FFFu + ((u >> 16) & 1u);   // round-to-nearest-even
  union { unsigned short s; bf16_t b; } c; c.s = (unsigned short)(r >> 16);
  return c.b;
}

// ---------------------------------------------------------------------------
// K1: dec 1x1 conv + softmax over R, then F_i = F * (1 + M_i). Writes fp32
// (for pool/deform) and bf16 (WMMA B operand of off1).
// ---------------------------------------------------------------------------
__global__ void decouple_kernel(const float* __restrict__ F,
                                const float* __restrict__ dec_w,
                                const float* __restrict__ dec_b,
                                float* __restrict__ Fi,
                                bf16_t* __restrict__ Fib,
                                int region) {
  int P = blockIdx.x * blockDim.x + threadIdx.x;
  if (P >= NPIX) return;
  int b = P / HW_, hw = P % HW_;
  const float* base = F + (size_t)b * C_ * HW_ + hw;
  float l0 = dec_b[0], l1 = dec_b[1];
  for (int c = 0; c < C_; ++c) {
    float v = base[(size_t)c * HW_];
    l0 += v * dec_w[c];
    l1 += v * dec_w[C_ + c];
  }
  float mx = fmaxf(l0, l1);
  float e0 = __expf(l0 - mx), e1 = __expf(l1 - mx);
  float m = (region == 0 ? e0 : e1) / (e0 + e1);
  float s = 1.0f + m;
  float*  of = Fi  + (size_t)b * C_ * HW_ + hw;
  bf16_t* ob = Fib + (size_t)b * C_ * HW_ + hw;
  for (int c = 0; c < C_; ++c) {
    float v = base[(size_t)c * HW_] * s;
    of[(size_t)c * HW_] = v;
    ob[(size_t)c * HW_] = f2bf(v);
  }
}

// ---------------------------------------------------------------------------
// Weight packer: OIHW fp32 -> bf16 A-fragments in the documented CDNA5 16-bit
// A-matrix 16x32 layout: per lane l (m=l&15, half=l>>4), element e: vgpr
// v=e>>1, k = (v<4 ? 2v : 16+2(v-4)) + 8*half + (e&1). K ordering is
// gk = cin*T + tap. Rows >= CoutReal are zero padding. One contiguous 32B
// v16bf load per lane per K-chunk in the GEMM.
// ---------------------------------------------------------------------------
template <int T>
__global__ void pack_a_kernel(const float* __restrict__ w, bf16_t* __restrict__ Apack,
                              int CoutReal, int Cin, int KC, int total) {
  int idx = blockIdx.x * blockDim.x + threadIdx.x;
  if (idx >= total) return;
  int e    = idx & 15;
  int lane = (idx >> 4) & 31;
  int kc   = (idx >> 9) % KC;
  int mt   = idx / (512 * KC);
  int m = lane & 15, half = lane >> 4;
  int v = e >> 1;
  int klocal = ((v < 4) ? (v * 2) : (16 + (v - 4) * 2)) + half * 8 + (e & 1);
  int gk = kc * 32 + klocal;
  int co = mt * 16 + m;
  int ci = gk / T, tap = gk % T;
  float val = (co < CoutReal) ? w[((size_t)co * Cin + ci) * T + tap] : 0.0f;
  Apack[idx] = f2bf(val);
}

// ---------------------------------------------------------------------------
// Conv-as-GEMM with v_wmma_f32_16x16x32_bf16, fully compile-time shaped.
//   T==9 : 3x3 conv, pad 1 (im2col on the fly)     T==1 : 1x1 conv
// Block: 32*MT threads; wave w owns channel tile [16w,16w+16); block owns
// NT*16 pixels. Per K-chunk: stage NT 32x16 B patches into LDS in B-fragment
// order (lane l: n=l&15, K=((e&8)<<1)+8*(l>>4)+(e&7)), one A-fragment load per wave reused
// across NT WMMAs. C/D layout: vgpr r, lane l -> m=8*(l>>4)+r, n=l&15.
// ---------------------------------------------------------------------------
template <int CIN, int T, int COUT, int MT, int NT, bool RELU>
__global__ __launch_bounds__(32 * MT)
void conv_wmma_kernel(const bf16_t* __restrict__ in,
                      const bf16_t* __restrict__ Apack,
                      const float*  __restrict__ bias,
                      bf16_t* __restrict__ out_bf,
                      float*  __restrict__ out_f,
                      int chStride) {
  constexpr int KC = CIN * T / 32;
  constexpr int NTHREADS = 32 * MT;
  __shared__ __align__(32) bf16_t Bf[NT * 512];
  const int tid  = threadIdx.x;
  const int wave = tid >> 5;
  const int lane = tid & 31;
  const int pbase = blockIdx.x * (16 * NT);
  const bf16_t bz = f2bf(0.0f);

  const bf16_t* aptr = Apack + ((size_t)wave * KC * 32 + lane) * 16;

  v8f acc[NT];
#pragma unroll
  for (int nt = 0; nt < NT; ++nt) acc[nt] = v8f{};

  for (int kc = 0; kc < KC; ++kc) {
    // Stage NT B-fragments (im2col on the fly), fragment-ordered in LDS.
#pragma unroll 4
    for (int idx = tid; idx < NT * 512; idx += NTHREADS) {
      int nt = idx >> 9;
      int l  = (idx >> 4) & 31;
      int e  = idx & 15;
      int n  = nt * 16 + (l & 15);
      int K  = ((e & 8) << 1) + ((l >> 4) << 3) + (e & 7);
      int gk = kc * 32 + K;
      int ci = gk / T, tap = gk % T;       // compile-time divisor
      int P = pbase + n;
      int b = P / HW_, hw = P % HW_, y = hw / W_, x = hw % W_;
      bf16_t val = bz;
      if (T == 9) {
        int yy = y + tap / 3 - 1, xx = x + tap % 3 - 1;
        if (yy >= 0 && yy < H_ && xx >= 0 && xx < W_)
          val = in[(((size_t)b * CIN + ci) * H_ + yy) * W_ + xx];
      } else {
        val = in[(((size_t)b * CIN + ci) * H_ + y) * W_ + x];
      }
      Bf[idx] = val;
    }
    __syncthreads();
    v16bf a = *(const v16bf*)(aptr + (size_t)kc * 32 * 16);
    if (kc + 1 < KC)  // prefetch next A chunk (global_prefetch_b8 path)
      __builtin_prefetch(aptr + (size_t)(kc + 1) * 32 * 16, 0, 3);
#pragma unroll
    for (int nt = 0; nt < NT; ++nt) {
      v16bf bb = *(const v16bf*)(Bf + nt * 512 + lane * 16);
      acc[nt] = __builtin_amdgcn_wmma_f32_16x16x32_bf16(
          false, a, false, bb, (short)0, acc[nt], false, false);
    }
    __syncthreads();
  }

  // Scatter C tiles.
#pragma unroll
  for (int nt = 0; nt < NT; ++nt) {
    int n = lane & 15;
    int P = pbase + nt * 16 + n;
    int pb = P / HW_, hw = P % HW_, y = hw / W_, x = hw % W_;
#pragma unroll
    for (int r = 0; r < 8; ++r) {
      int m = ((lane >> 4) << 3) + r;
      int co = wave * 16 + m;
      if (co >= COUT) continue;
      float v = acc[nt][r] + (bias ? bias[co] : 0.0f);
      if (RELU) v = fmaxf(v, 0.0f);
      size_t oidx = (((size_t)pb * chStride + co) * H_ + y) * W_ + x;
      if (out_f)  out_f[oidx]  = v;
      if (out_bf) out_bf[oidx] = f2bf(v);
    }
  }
}

// ---------------------------------------------------------------------------
// AKG branch (tiny): 8x8 mean pool 56->7, depthwise 3x3 + relu + GAP, 1x1
// head C -> C*9 with tanh.
// ---------------------------------------------------------------------------
__global__ void pool_kernel(const float* __restrict__ Fi, float* __restrict__ pooled) {
  int idx = blockIdx.x * blockDim.x + threadIdx.x;
  if (idx >= B_ * C_ * KD * KD) return;
  int q = idx % (KD * KD);
  int bc = idx / (KD * KD);
  int qy = q / KD, qx = q % KD;
  const float* base = Fi + (size_t)bc * HW_;
  float s = 0.0f;
  for (int yy = 0; yy < 8; ++yy)
    for (int xx = 0; xx < 8; ++xx)
      s += base[(qy * 8 + yy) * W_ + qx * 8 + xx];
  pooled[idx] = s * (1.0f / 64.0f);
}

__global__ void akg_dw_kernel(const float* __restrict__ pooled,
                              const float* __restrict__ dww,
                              float* __restrict__ dvec) {
  int bc = blockIdx.x * blockDim.x + threadIdx.x;
  if (bc >= B_ * C_) return;
  int c = bc % C_;
  const float* p = pooled + (size_t)bc * KD * KD;
  const float* w = dww + (size_t)c * 9;
  float s = 0.0f;
  for (int y = 0; y < KD; ++y)
    for (int x = 0; x < KD; ++x) {
      float a = 0.0f;
      for (int t = 0; t < 9; ++t) {
        int yy = y + t / 3 - 1, xx = x + t % 3 - 1;
        if (yy >= 0 && yy < KD && xx >= 0 && xx < KD) a += p[yy * KD + xx] * w[t];
      }
      s += fmaxf(a, 0.0f);
    }
  dvec[bc] = s * (1.0f / (KD * KD));
}

__global__ void akg_head_kernel(const float* __restrict__ dvec,
                                const float* __restrict__ hw,
                                const float* __restrict__ hb,
                                float* __restrict__ Ki) {
  int idx = blockIdx.x * blockDim.x + threadIdx.x;
  if (idx >= B_ * C_ * 9) return;
  int o = idx % (C_ * 9), b = idx / (C_ * 9);
  const float* dv = dvec + (size_t)b * C_;
  const float* wr = hw + (size_t)o * C_;
  float s = hb[o];
  for (int c = 0; c < C_; ++c) s += dv[c] * wr[c];
  Ki[idx] = tanhf(s);
}

// ---------------------------------------------------------------------------
// Per-sample dynamic depthwise deformable 3x3 (the gather-bound core).
// Writes bf16 straight into the concat buffer (B, 2C, H, W) for pw2's WMMA.
// ---------------------------------------------------------------------------
__global__ void deform_kernel(const float* __restrict__ Fi,
                              const float* __restrict__ off,
                              const float* __restrict__ Ki,
                              bf16_t* __restrict__ concat, int chOff) {
  size_t idx = (size_t)blockIdx.x * blockDim.x + threadIdx.x;
  if (idx >= (size_t)B_ * C_ * HW_) return;
  int b  = (int)(idx / ((size_t)C_ * HW_));
  int rm = (int)(idx % ((size_t)C_ * HW_));
  int c  = rm / HW_, hw = rm % HW_;
  int y = hw / W_, x = hw % W_;
  const float* img = Fi  + ((size_t)b * C_ + c) * HW_;
  const float* kp  = Ki  + ((size_t)b * C_ + c) * 9;
  const float* pb  = off + (size_t)b * OFFC * HW_ + hw;

  float s = 0.0f;
#pragma unroll
  for (int t = 0; t < 9; ++t) {
    float py = (float)(y - 1 + t / 3) + pb[(size_t)(2 * t) * HW_];
    float px = (float)(x - 1 + t % 3) + pb[(size_t)(2 * t + 1) * HW_];
    float y0f = floorf(py), x0f = floorf(px);
    float wy1 = py - y0f, wx1 = px - x0f;
    float wy0 = 1.0f - wy1, wx0 = 1.0f - wx1;
    int iy0 = (int)y0f, ix0 = (int)x0f;
    int iy1 = iy0 + 1, ix1 = ix0 + 1;
    float g00 = (iy0 >= 0 && iy0 < H_ && ix0 >= 0 && ix0 < W_) ? img[iy0 * W_ + ix0] : 0.0f;
    float g01 = (iy0 >= 0 && iy0 < H_ && ix1 >= 0 && ix1 < W_) ? img[iy0 * W_ + ix1] : 0.0f;
    float g10 = (iy1 >= 0 && iy1 < H_ && ix0 >= 0 && ix0 < W_) ? img[iy1 * W_ + ix0] : 0.0f;
    float g11 = (iy1 >= 0 && iy1 < H_ && ix1 >= 0 && ix1 < W_) ? img[iy1 * W_ + ix1] : 0.0f;
    float v = g00 * wy0 * wx0 + g01 * wy0 * wx1 + g10 * wy1 * wx0 + g11 * wy1 * wx1;
    s += kp[t] * v;
  }
  concat[(((size_t)b * (2 * C_) + chOff + c) * HW_) + hw] = f2bf(s);
}

// ---------------------------------------------------------------------------
// Training-mode BatchNorm: per-channel reduction (no atomics), then apply.
// ---------------------------------------------------------------------------
__global__ void bn_reduce_kernel(const float* __restrict__ O,
                                 const float* __restrict__ gamma,
                                 const float* __restrict__ beta,
                                 float* __restrict__ scale,
                                 float* __restrict__ shift) {
  __shared__ float ssum[256];
  __shared__ float ssq[256];
  int ch = blockIdx.x;
  int tid = threadIdx.x;
  float s = 0.0f, q = 0.0f;
  for (int j = tid; j < NPIX; j += blockDim.x) {
    int b = j / HW_, hw = j % HW_;
    float v = O[((size_t)b * C_ + ch) * HW_ + hw];
    s += v; q += v * v;
  }
  ssum[tid] = s; ssq[tid] = q;
  __syncthreads();
  for (int st = 128; st > 0; st >>= 1) {
    if (tid < st) { ssum[tid] += ssum[tid + st]; ssq[tid] += ssq[tid + st]; }
    __syncthreads();
  }
  if (tid == 0) {
    float mean = ssum[0] / (float)NPIX;
    float var  = ssq[0] / (float)NPIX - mean * mean;
    float inv  = rsqrtf(var + EPS_BN);
    float sc = gamma[ch] * inv;
    scale[ch] = sc;
    shift[ch] = beta[ch] - mean * sc;
  }
}

__global__ void bn_apply_kernel(float* __restrict__ O,
                                const float* __restrict__ scale,
                                const float* __restrict__ shift) {
  size_t idx = (size_t)blockIdx.x * blockDim.x + threadIdx.x;
  if (idx >= (size_t)B_ * C_ * HW_) return;
  int ch = (int)((idx / HW_) % C_);
  O[idx] = O[idx] * scale[ch] + shift[ch];
}

// ---------------------------------------------------------------------------
// Host-side orchestration (graph-capturable: stream launches only).
// ---------------------------------------------------------------------------
extern "C" void kernel_launch(void* const* d_in, const int* in_sizes, int n_in,
                              void* d_out, int out_size, void* d_ws, size_t ws_size,
                              hipStream_t stream) {
  (void)in_sizes; (void)n_in; (void)out_size; (void)ws_size;

  const float* F_c        = (const float*)d_in[0];
  const float* dec_w      = (const float*)d_in[1];
  const float* dec_b      = (const float*)d_in[2];
  const float* off1_w     = (const float*)d_in[3];
  const float* off1_b     = (const float*)d_in[4];
  const float* off2_w     = (const float*)d_in[5];
  const float* off2_b     = (const float*)d_in[6];
  const float* akg_dw_w   = (const float*)d_in[7];
  const float* akg_head_w = (const float*)d_in[8];
  const float* akg_head_b = (const float*)d_in[9];
  const float* pw2_w      = (const float*)d_in[10];
  const float* pw2_b      = (const float*)d_in[11];
  const float* bn_gamma   = (const float*)d_in[12];
  const float* bn_beta    = (const float*)d_in[13];

  // GEMM shapes.
  constexpr int KC1 = C_ * 9 / 32;      // 72
  constexpr int KC2 = HID * 9 / 32;     // 18
  constexpr int KCP = 2 * C_ / 32;      // 16
  constexpr int MT1 = HID / 16;         // 4 waves
  constexpr int MT2 = 2;                // 18 -> padded to 32 rows
  constexpr int MTP = C_ / 16;          // 16 waves
  constexpr int NT  = 2;                // 32 pixels / block

  // Workspace layout (bytes, 256B aligned).
  constexpr size_t SZ_FI    = (size_t)NPIX * C_ * 4;        // 25.7 MB
  constexpr size_t SZ_FIB   = (size_t)NPIX * C_ * 2;        // 12.8 MB
  constexpr size_t SZ_HBF   = (size_t)NPIX * HID * 2;       // 3.2 MB
  constexpr size_t SZ_P     = (size_t)NPIX * OFFC * 4;      // 1.8 MB
  constexpr size_t SZ_POOL  = (size_t)B_ * C_ * KD * KD * 4;
  constexpr size_t SZ_DVEC  = (size_t)B_ * C_ * 4;
  constexpr size_t SZ_KI    = (size_t)B_ * C_ * 9 * 4;
  constexpr size_t SZ_CAT   = (size_t)NPIX * 2 * C_ * 2;    // 25.7 MB
  constexpr size_t SZ_A1 = (size_t)MT1 * KC1 * 512 * 2;
  constexpr size_t SZ_A2 = (size_t)MT2 * KC2 * 512 * 2;
  constexpr size_t SZ_AP = (size_t)MTP * KCP * 512 * 2;

  auto al = [](size_t x) { return (x + 255) & ~(size_t)255; };
  char* ws = (char*)d_ws;
  size_t o = 0;
  float*  Fi     = (float*) (ws + o); o += al(SZ_FI);
  bf16_t* Fib    = (bf16_t*)(ws + o); o += al(SZ_FIB);
  bf16_t* h_bf   = (bf16_t*)(ws + o); o += al(SZ_HBF);
  float*  p_buf  = (float*) (ws + o); o += al(SZ_P);
  float*  pooled = (float*) (ws + o); o += al(SZ_POOL);
  float*  dvec   = (float*) (ws + o); o += al(SZ_DVEC);
  float*  Ki     = (float*) (ws + o); o += al(SZ_KI);
  bf16_t* concat = (bf16_t*)(ws + o); o += al(SZ_CAT);
  bf16_t* A1     = (bf16_t*)(ws + o); o += al(SZ_A1);
  bf16_t* A2     = (bf16_t*)(ws + o); o += al(SZ_A2);
  bf16_t* AP     = (bf16_t*)(ws + o); o += al(SZ_AP);
  float*  bnsc   = (float*) (ws + o); o += al(256 * 4);
  float*  bnsh   = (float*) (ws + o); o += al(256 * 4);

  float* O = (float*)d_out;

  const int nblocks = NPIX / (16 * NT);  // 784 blocks per GEMM

  // Pack pw2 weights once (region-independent).
  {
    int total = MTP * KCP * 512;
    pack_a_kernel<1><<<(total + 255) / 256, 256, 0, stream>>>(
        pw2_w, AP, C_, 2 * C_, KCP, total);
  }

  for (int i = 0; i < R_; ++i) {
    // F_i = F * (1 + softmax(dec(F))[i])
    decouple_kernel<<<(NPIX + 255) / 256, 256, 0, stream>>>(
        F_c, dec_w, dec_b, Fi, Fib, i);

    // off1: 3x3 conv C->HID, relu (WMMA)
    {
      int total = MT1 * KC1 * 512;
      pack_a_kernel<9><<<(total + 255) / 256, 256, 0, stream>>>(
          off1_w + (size_t)i * HID * C_ * 9, A1, HID, C_, KC1, total);
      conv_wmma_kernel<C_, 9, HID, MT1, NT, true>
          <<<nblocks, 32 * MT1, 0, stream>>>(
              Fib, A1, off1_b + (size_t)i * HID, h_bf, nullptr, HID);
    }
    // off2: 3x3 conv HID->18 (WMMA), fp32 offsets out
    {
      int total = MT2 * KC2 * 512;
      pack_a_kernel<9><<<(total + 255) / 256, 256, 0, stream>>>(
          off2_w + (size_t)i * OFFC * HID * 9, A2, OFFC, HID, KC2, total);
      conv_wmma_kernel<HID, 9, OFFC, MT2, NT, false>
          <<<nblocks, 32 * MT2, 0, stream>>>(
              h_bf, A2, off2_b + (size_t)i * OFFC, nullptr, p_buf, OFFC);
    }

    // AKG branch
    pool_kernel<<<(B_ * C_ * KD * KD + 255) / 256, 256, 0, stream>>>(Fi, pooled);
    akg_dw_kernel<<<(B_ * C_ + 255) / 256, 256, 0, stream>>>(
        pooled, akg_dw_w + (size_t)i * C_ * 9, dvec);
    akg_head_kernel<<<(B_ * C_ * 9 + 255) / 256, 256, 0, stream>>>(
        dvec, akg_head_w + (size_t)i * C_ * 9 * C_,
        akg_head_b + (size_t)i * C_ * 9, Ki);

    // Deformable dynamic depthwise conv -> concat channels [i*C, i*C+C)
    {
      size_t total = (size_t)B_ * C_ * HW_;
      deform_kernel<<<(unsigned)((total + 255) / 256), 256, 0, stream>>>(
          Fi, p_buf, Ki, concat, i * C_);
    }
  }

  // pw2: 1x1 conv 2C->C (WMMA), fp32 O into d_out
  conv_wmma_kernel<2 * C_, 1, C_, MTP, NT, false>
      <<<nblocks, 32 * MTP, 0, stream>>>(
          concat, AP, pw2_b, nullptr, O, C_);

  // BatchNorm (batch stats)
  bn_reduce_kernel<<<C_, 256, 0, stream>>>(O, bn_gamma, bn_beta, bnsc, bnsh);
  {
    size_t total = (size_t)B_ * C_ * HW_;
    bn_apply_kernel<<<(unsigned)((total + 255) / 256), 256, 0, stream>>>(O, bnsc, bnsh);
  }
}
